// TransformerBlock_41214506172529
// MI455X (gfx1250) — matrix operand order repaired and checked by reading, not compiled, not run on hardware
//
#include <hip/hip_runtime.h>
#include <math.h>

// ---------------- constants (match reference) ----------------
#define BB   2
#define NN   2048
#define CC   1024
#define HH   16
#define HD   64
#define HID  4096
#define MM   (BB * NN)          // 4096 rows
#define SCALE 0.125f            // 64^-0.5

typedef __attribute__((ext_vector_type(16))) __bf16 bf16x16;
typedef __attribute__((ext_vector_type(8)))  __bf16 bf16x8;
typedef __attribute__((ext_vector_type(8)))  float  f32x8;

// A-slab staged in LDS with padded row stride to spread banks
#define ASTRIDE 40   // bf16 elements per LDS row (64B data + 16B pad)

#if defined(__has_builtin)
#if __has_builtin(__builtin_amdgcn_sched_barrier)
#define SCHED_FENCE() __builtin_amdgcn_sched_barrier(0)
#endif
#endif
#ifndef SCHED_FENCE
#define SCHED_FENCE() do {} while (0)
#endif

// ---------------- CDNA5 helpers ----------------
__device__ __forceinline__ uint32_t lds_off32(const void* p) {
    // generic -> LDS address space, then take the 32-bit LDS byte offset
    return (uint32_t)(uintptr_t)(__attribute__((address_space(3))) const void*)p;
}

// per-lane async copy of 16B global -> LDS (tracked by ASYNCcnt)
__device__ __forceinline__ void async_copy_b128(uint32_t lds_off, uint64_t gaddr) {
    asm volatile("global_load_async_to_lds_b128 %0, %1, off"
                 :: "v"(lds_off), "v"(gaddr) : "memory");
}
__device__ __forceinline__ void wait_asynccnt0() {
    asm volatile("s_wait_asynccnt 0" ::: "memory");
}

// ---------------- fragment helpers ----------------
// A fragment: 16x32 bf16 (M x K).  Lane l (l2=l&15, lh=l>>4) holds row
// mbase+l2, K chunks {k+lh*8 .. +7} and {k+16+lh*8 .. +7}  (ISA 7.12.2).
__device__ __forceinline__ bf16x16 load_a_frag(const __bf16* A, int lda,
                                               int mbase, int k, int l2, int lh) {
    const __bf16* p = A + (size_t)(mbase + l2) * lda + k + lh * 8;
    union { bf16x16 v; bf16x8 h[2]; } u;
    u.h[0] = *(const bf16x8*)(p);
    u.h[1] = *(const bf16x8*)(p + 16);
    return u.v;
}

// A fragment out of the LDS-staged slab (row-major, stride ASTRIDE)
__device__ __forceinline__ bf16x16 lds_a_frag(const __bf16* slab,
                                              int mrel, int l2, int lh) {
    const __bf16* p = slab + (size_t)(mrel + l2) * ASTRIDE + lh * 8;
    union { bf16x16 v; bf16x8 h[2]; } u;
    u.h[0] = *(const bf16x8*)(p);
    u.h[1] = *(const bf16x8*)(p + 16);
    return u.v;
}

// B fragment: 32x16 bf16 (K x N) where memory is row-major [N, K] (i.e. B^T).
// Lane l holds column n = nbase+l2, K chunks {k+lh*8 .. +7} and {k+16+lh*8 .. +7}.
__device__ __forceinline__ bf16x16 load_b_frag(const __bf16* Bm, int ldb,
                                               int nbase, int k, int l2, int lh) {
    const __bf16* p = Bm + (size_t)(nbase + l2) * ldb + k + lh * 8;
    union { bf16x16 v; bf16x8 h[2]; } u;
    u.h[0] = *(const bf16x8*)(p);
    u.h[1] = *(const bf16x8*)(p + 16);
    return u.v;
}

__device__ __forceinline__ f32x8 wmma_bf16(bf16x16 a, bf16x16 b, f32x8 c) {
    return __builtin_amdgcn_wmma_f32_16x16x32_bf16(false, a, false, b,
                                                   (short)0, c, false, false);
}

__device__ __forceinline__ float gelu_exact(float x) {
    return 0.5f * x * (1.0f + erff(x * 0.70710678118654752f));
}

// ---------------- fp32 -> bf16 convert ----------------
__global__ __launch_bounds__(256) void k_cvt(const float* __restrict__ in,
                                             __bf16* __restrict__ out, int n) {
    int i = blockIdx.x * 256 + threadIdx.x;
    if (i < n) out[i] = (__bf16)in[i];
}

// ---------------- WMMA GEMM: D[M,N] = A[M,K] @ B[N,K]^T (+bias, +act) -------
// Block = 4 waves covering 4 consecutive 64x64 N-tiles of one 64-row M slab.
// The shared 64x32 A slab is async-staged into LDS (double buffered) and kept
// in flight across the loads + 16-WMMA burst; B fragments are direct clause'd
// global loads (L2 resident).
__global__ __launch_bounds__(128) void k_gemm(const __bf16* __restrict__ A,
                                              const __bf16* __restrict__ Bm,
                                              const float*  __restrict__ bias,
                                              float*   outF, __bf16* outB,
                                              int M, int N, int K, int ldout,
                                              int act) {
    const int tid  = threadIdx.x;
    const int lane = tid & 31;
    const int wave = tid >> 5;
    const int l2 = lane & 15, lh = lane >> 4;

    const int ntn  = N >> 6;                  // 64-wide N tiles per row (mult of 4)
    const int tile = blockIdx.x * 4 + wave;
    const int mbase = (tile / ntn) * 64;
    const int nbase = (tile % ntn) * 64;

    __shared__ __bf16 abuf[2][64 * ASTRIDE];
    const uint32_t lbase[2] = { lds_off32(&abuf[0][0]), lds_off32(&abuf[1][0]) };

    // stage helper: thread t copies 32B of row t/2 (two async b128)
    const int srow = tid >> 1, shalf = tid & 1;
    const size_t g_row_off = (size_t)(mbase + srow) * K + shalf * 16;
    const uint32_t l_row_off = (uint32_t)((srow * ASTRIDE + shalf * 16) * 2);

    // prologue: stage k=0 slab
    {
        uint64_t ga = (uint64_t)(uintptr_t)(A + g_row_off);
        async_copy_b128(lbase[0] + l_row_off,      ga);
        async_copy_b128(lbase[0] + l_row_off + 16, ga + 16);
    }
    wait_asynccnt0();
    __syncthreads();

    f32x8 acc[4][4] = {};
    int cur = 0;

    for (int k = 0; k < K; k += 32) {
        // 1) kick off the async stage of the NEXT A slab first, so the DMA is
        //    in flight across this iteration's loads and WMMA burst.  Safe:
        //    buffer cur^1 was last read one iteration ago and the barrier at
        //    the end of the previous iteration ordered all its ds reads.
        if (k + 32 < K) {
            uint64_t ga = (uint64_t)(uintptr_t)(A + g_row_off + (k + 32));
            async_copy_b128(lbase[cur ^ 1] + l_row_off,      ga);
            async_copy_b128(lbase[cur ^ 1] + l_row_off + 16, ga + 16);
            // L2 prefetch of the B slab after next -> global_prefetch_b8
            __builtin_prefetch(Bm + (size_t)(nbase + l2) * K + k + 64, 0, 1);
        }

        // 2) fragment loads for this k-step
        bf16x16 a[4], b[4];
        const __bf16* slab = &abuf[cur][0];
#pragma unroll
        for (int i = 0; i < 4; ++i)
            a[i] = lds_a_frag(slab, 16 * i, l2, lh);
#pragma unroll
        for (int j = 0; j < 4; ++j)
            b[j] = load_b_frag(Bm, K, nbase + 16 * j, k, l2, lh);

        // 3) 16-WMMA burst
#pragma unroll
        for (int i = 0; i < 4; ++i)
#pragma unroll
            for (int j = 0; j < 4; ++j)
                acc[i][j] = wmma_bf16(a[i], b[j], acc[i][j]);

        // 4) only now drain the async copies; the sched fence keeps the
        //    scheduler from hoisting the wait above the (pure) WMMAs.
        SCHED_FENCE();
        wait_asynccnt0();
        __syncthreads();
        cur ^= 1;
    }

    // Epilogue.  C layout: vgpr r, lane l -> row r + 8*lh, col l2 (ISA 7.12.2).
#pragma unroll
    for (int i = 0; i < 4; ++i)
#pragma unroll
        for (int j = 0; j < 4; ++j) {
            const int col = nbase + 16 * j + l2;
            float bcol = bias ? bias[col] : 0.0f;
#pragma unroll
            for (int r = 0; r < 8; ++r) {
                const int row = mbase + 16 * i + r + 8 * lh;
                float v = acc[i][j][r] + bcol;
                if (act == 1) v = gelu_exact(v);
                const size_t off = (size_t)row * ldout + col;
                if (outF) outF[off] = v;
                if (outB) outB[off] = (__bf16)v;
            }
        }
}

// ---------------- QKV repack ----------------
// qkv_tmp [B*N, 3C] -> Q,K [B*H, N, HD] and V^T [B*H, HD, N]
__global__ __launch_bounds__(256) void k_qkv_repack(const __bf16* __restrict__ t,
                                                    __bf16* __restrict__ Q,
                                                    __bf16* __restrict__ K,
                                                    __bf16* __restrict__ Vt) {
    int i = blockIdx.x * 256 + threadIdx.x;          // over B*N*C
    if (i >= MM * CC) return;
    const int row = i / CC;        // b*N + n
    const int col = i % CC;        // h*HD + d
    const int b = row / NN, n = row % NN;
    const int h = col / HD, d = col % HD;
    const size_t bh = (size_t)b * HH + h;
    const size_t src = (size_t)row * (3 * CC) + col;
    Q [bh * NN * HD + (size_t)n * HD + d] = t[src];
    K [bh * NN * HD + (size_t)n * HD + d] = t[src + CC];
    Vt[bh * HD * NN + (size_t)d * NN + n] = t[src + 2 * CC];
}

// ---------------- flash attention (one wave = 16 query rows) ----------------
__global__ __launch_bounds__(128) void k_attn(const __bf16* __restrict__ Q,
                                              const __bf16* __restrict__ Km,
                                              const __bf16* __restrict__ Vt,
                                              __bf16* __restrict__ ctx) {
    const int qblocks = NN / 64;
    const int bh   = blockIdx.x / qblocks;
    const int qblk = blockIdx.x % qblocks;
    const int lane = threadIdx.x & 31;
    const int wave = threadIdx.x >> 5;
    const int l2 = lane & 15, lh = lane >> 4;
    const int qbase = qblk * 64 + wave * 16;

    const __bf16* Qp = Q  + (size_t)bh * NN * HD;
    const __bf16* Kp = Km + (size_t)bh * NN * HD;
    const __bf16* Vp = Vt + (size_t)bh * HD * NN;

    __shared__ __bf16 pbuf[4][16 * 32];
    __bf16* P = pbuf[wave];

    const bf16x16 a0 = load_a_frag(Qp, HD, qbase, 0,  l2, lh);
    const bf16x16 a1 = load_a_frag(Qp, HD, qbase, 32, l2, lh);

    f32x8 o[4] = {};
    float m_i[8], l_i[8];
#pragma unroll
    for (int r = 0; r < 8; ++r) { m_i[r] = -3.0e38f; l_i[r] = 0.0f; }

    for (int j = 0; j < NN; j += 32) {
        // S = Q @ K^T for 32 keys (two 16-wide C tiles, hd-depth 64)
        f32x8 s0 = {}, s1 = {};
        s0 = wmma_bf16(a0, load_b_frag(Kp, HD, j,      0,  l2, lh), s0);
        s0 = wmma_bf16(a1, load_b_frag(Kp, HD, j,      32, l2, lh), s0);
        s1 = wmma_bf16(a0, load_b_frag(Kp, HD, j + 16, 0,  l2, lh), s1);
        s1 = wmma_bf16(a1, load_b_frag(Kp, HD, j + 16, 32, l2, lh), s1);

        // online softmax; row stats uniform within each 16-lane half
#pragma unroll
        for (int r = 0; r < 8; ++r) {
            float x0 = s0[r] * SCALE, x1 = s1[r] * SCALE;
            float mx = fmaxf(x0, x1);
#pragma unroll
            for (int d = 1; d < 16; d <<= 1) mx = fmaxf(mx, __shfl_xor(mx, d, 32));
            const float mnew = fmaxf(m_i[r], mx);
            const float sf = __expf(m_i[r] - mnew);
            m_i[r] = mnew;
            const float p0 = __expf(x0 - mnew);
            const float p1 = __expf(x1 - mnew);
            float rs = p0 + p1;
#pragma unroll
            for (int d = 1; d < 16; d <<= 1) rs += __shfl_xor(rs, d, 32);
            l_i[r] = l_i[r] * sf + rs;
#pragma unroll
            for (int t = 0; t < 4; ++t) o[t][r] *= sf;
            // stash P (bf16) row-major [16 x 32] for the LDS transpose
            P[(r + 8 * lh) * 32 + l2]      = (__bf16)p0;
            P[(r + 8 * lh) * 32 + 16 + l2] = (__bf16)p1;
        }
        __builtin_amdgcn_wave_barrier();

        // reload P as an A fragment (16 queries x 32 keys)
        bf16x16 pa;
        {
            union { bf16x16 v; bf16x8 h[2]; } u;
            const __bf16* pp = P + l2 * 32 + lh * 8;
            u.h[0] = *(const bf16x8*)(pp);
            u.h[1] = *(const bf16x8*)(pp + 16);
            pa = u.v;
        }
        __builtin_amdgcn_wave_barrier();

        // O += P @ V   (V^T rows give contiguous B fragments)
#pragma unroll
        for (int t = 0; t < 4; ++t)
            o[t] = wmma_bf16(pa, load_b_frag(Vp, NN, t * 16, j, l2, lh), o[t]);
    }

    // normalize and scatter ctx [B, N, C], col = h*HD + ...
    const int b = bh / HH, h = bh % HH;
#pragma unroll
    for (int r = 0; r < 8; ++r) {
        const float inv = 1.0f / l_i[r];
        const int q = qbase + r + 8 * lh;
#pragma unroll
        for (int t = 0; t < 4; ++t) {
            const int col = h * HD + t * 16 + l2;
            ctx[((size_t)b * NN + q) * CC + col] = (__bf16)(o[t][r] * inv);
        }
    }
}

// ---------------- fused residual + LayerNorm ----------------
// out = resid + LN(xin) * w + b ; one block per row of 1024
__global__ __launch_bounds__(256) void k_ln_resid(const float* __restrict__ xin,
                                                  const float* __restrict__ resid,
                                                  const float* __restrict__ w,
                                                  const float* __restrict__ b,
                                                  float* outF, __bf16* outB) {
    const int row = blockIdx.x;
    const int tid = threadIdx.x;
    const int lane = tid & 31, wave = tid >> 5;
    __shared__ float s_sum[8], s_sq[8];

    float sum = 0.0f, sq = 0.0f;
    for (int i = tid; i < CC; i += 256) {
        float v = xin[(size_t)row * CC + i];
        sum += v; sq += v * v;
    }
#pragma unroll
    for (int d = 1; d < 32; d <<= 1) {
        sum += __shfl_xor(sum, d, 32);
        sq  += __shfl_xor(sq,  d, 32);
    }
    if (lane == 0) { s_sum[wave] = sum; s_sq[wave] = sq; }
    __syncthreads();
    sum = 0.0f; sq = 0.0f;
#pragma unroll
    for (int i = 0; i < 8; ++i) { sum += s_sum[i]; sq += s_sq[i]; }
    const float mu = sum * (1.0f / CC);
    const float var = sq * (1.0f / CC) - mu * mu;
    const float rs = rsqrtf(var + 1e-5f);

    for (int i = tid; i < CC; i += 256) {
        const size_t off = (size_t)row * CC + i;
        float v = xin[off];
        float y = resid[off] + (v - mu) * rs * w[i] + b[i];
        if (outF) outF[off] = y;
        if (outB) outB[off] = (__bf16)y;
    }
}

// ---------------- host orchestration ----------------
extern "C" void kernel_launch(void* const* d_in, const int* in_sizes, int n_in,
                              void* d_out, int out_size, void* d_ws, size_t ws_size,
                              hipStream_t stream) {
    const float* x      = (const float*)d_in[0];
    const float* qkv_w  = (const float*)d_in[1];
    const float* proj_w = (const float*)d_in[2];
    const float* proj_b = (const float*)d_in[3];
    const float* ln1_w  = (const float*)d_in[4];
    const float* ln1_b  = (const float*)d_in[5];
    const float* fc1_w  = (const float*)d_in[6];
    const float* fc1_b  = (const float*)d_in[7];
    const float* fc2_w  = (const float*)d_in[8];
    const float* fc2_b  = (const float*)d_in[9];
    const float* ln2_w  = (const float*)d_in[10];
    const float* ln2_b  = (const float*)d_in[11];
    float* out = (float*)d_out;

    char* ws = (char*)d_ws;
    size_t off = 0;
    auto alloc = [&](size_t bytes) -> void* {
        void* p = ws + off;
        off += (bytes + 255) & ~(size_t)255;
        return p;
    };

    __bf16* xb      = (__bf16*)alloc((size_t)MM * CC * 2);
    __bf16* wqkv    = (__bf16*)alloc((size_t)3 * CC * CC * 2);
    __bf16* wproj   = (__bf16*)alloc((size_t)CC * CC * 2);
    __bf16* wfc1    = (__bf16*)alloc((size_t)HID * CC * 2);
    __bf16* wfc2    = (__bf16*)alloc((size_t)CC * HID * 2);
    __bf16* qkv_tmp = (__bf16*)alloc((size_t)MM * 3 * CC * 2);
    __bf16* Qb      = (__bf16*)alloc((size_t)BB * HH * NN * HD * 2);
    __bf16* Kb      = (__bf16*)alloc((size_t)BB * HH * NN * HD * 2);
    __bf16* Vtb     = (__bf16*)alloc((size_t)BB * HH * HD * NN * 2);
    __bf16* ctxb    = (__bf16*)alloc((size_t)MM * CC * 2);
    float*  attnout = (float*) alloc((size_t)MM * CC * 4);
    float*  res1    = (float*) alloc((size_t)MM * CC * 4);
    __bf16* res1b   = (__bf16*)alloc((size_t)MM * CC * 2);
    __bf16* hb      = (__bf16*)alloc((size_t)MM * HID * 2);
    float*  mlp     = (float*) alloc((size_t)MM * CC * 4);

    auto cvt = [&](const float* src, __bf16* dst, int n) {
        k_cvt<<<(n + 255) / 256, 256, 0, stream>>>(src, dst, n);
    };
    cvt(x,      xb,    MM * CC);
    cvt(qkv_w,  wqkv,  3 * CC * CC);
    cvt(proj_w, wproj, CC * CC);
    cvt(fc1_w,  wfc1,  HID * CC);
    cvt(fc2_w,  wfc2,  CC * HID);

    auto gemm = [&](const __bf16* A, const __bf16* Bm, const float* bias,
                    float* oF, __bf16* oB, int M, int N, int K, int act) {
        int tiles = (M / 64) * (N / 64);   // always a multiple of 4 here
        k_gemm<<<tiles / 4, 128, 0, stream>>>(A, Bm, bias, oF, oB,
                                              M, N, K, N, act);
    };

    // 1) QKV projection
    gemm(xb, wqkv, nullptr, nullptr, qkv_tmp, MM, 3 * CC, CC, 0);
    // 2) repack to per-head Q, K, V^T
    k_qkv_repack<<<(MM * CC + 255) / 256, 256, 0, stream>>>(qkv_tmp, Qb, Kb, Vtb);
    // 3) flash attention
    k_attn<<<BB * HH * (NN / 64), 128, 0, stream>>>(Qb, Kb, Vtb, ctxb);
    // 4) output projection (+bias)
    gemm(ctxb, wproj, proj_b, attnout, nullptr, MM, CC, CC, 0);
    // 5) out = attnout + LN(attnout)
    k_ln_resid<<<MM, 256, 0, stream>>>(attnout, attnout, ln1_w, ln1_b, res1, res1b);
    // 6) fc1 + GELU
    gemm(res1b, wfc1, fc1_b, nullptr, hb, MM, HID, CC, 1);
    // 7) fc2
    gemm(hb, wfc2, fc2_b, mlp, nullptr, MM, CC, HID, 0);
    // 8) final = res1 + LN(mlp)
    k_ln_resid<<<MM, 256, 0, stream>>>(mlp, res1, ln2_w, ln2_b, out, nullptr);
}
